// TransformerEncoder_4303557230769
// MI455X (gfx1250) — compile-verified, matrix operand order checked by reading, not run on hardware
//
#include <hip/hip_runtime.h>
#include <hip/hip_bf16.h>
#include <math.h>

typedef __bf16 bf16;
typedef __attribute__((ext_vector_type(16))) __bf16 v16bf;
typedef __attribute__((ext_vector_type(8)))  float  v8f;

#define S_  1024
#define D_  1024
#define H_  16
#define HD_ 64
#define F_  4096
#define L_  8

// ---------- helpers ----------

__device__ __forceinline__ bf16 f2bf(float f) {
  union { float f; unsigned u; } v; v.f = f;
  unsigned r = v.u + 0x7FFFu + ((v.u >> 16) & 1u);   // round-to-nearest-even
  unsigned short h = (unsigned short)(r >> 16);
  return __builtin_bit_cast(bf16, h);
}

__device__ __forceinline__ float gelu_tanh(float x) {
  float x3 = x * x * x;
  return 0.5f * x * (1.0f + tanhf(0.7978845608028654f * (x + 0.044715f * x3)));
}

// CDNA5 async copy: global -> LDS, 8 bytes per lane, tracked by ASYNCcnt.
__device__ __forceinline__ void async_b64(void* lds, const void* gaddr) {
  unsigned lds_off = (unsigned)(unsigned long long)(uintptr_t)lds;  // addr[31:0] = LDS offset
  asm volatile("global_load_async_to_lds_b64 %0, %1, off"
               :: "v"(lds_off), "v"((unsigned long long)(uintptr_t)gaddr)
               : "memory");
}
__device__ __forceinline__ void async_wait0() {
  asm volatile("s_wait_asynccnt 0x0" ::: "memory");
}

// Load a 16x32 bf16 WMMA fragment (A-layout; B uses the same layout when the
// tile is stored K-transposed, i.e. [n][k]).  ISA 7.12.2:
//   lanes 0-15 : row M = lane,    K = {0..7, 16..23}
//   lanes 16-31: row M = lane-16, K = {8..15, 24..31}
__device__ __forceinline__ v16bf load_frag(const bf16* p, int ld) {
  int lane = threadIdx.x & 31;
  int r = lane & 15;
  int h = lane >> 4;
  const bf16* q = p + r * ld + h * 8;
  v16bf f;
#pragma unroll
  for (int j = 0; j < 8; ++j) { f[j] = q[j]; f[8 + j] = q[16 + j]; }
  return f;
}

// ---------- GEMM: C[S x N] = A(bf16)[S x K] @ B(f32->bf16)[K x N] + bias ----------
// 256 threads, 8 waves, 128x128 block tile, K step 32, double-buffered LDS,
// async A copies + staged B loads issued before the WMMA block.

__global__ __launch_bounds__(256) void gemm_bf16_kernel(
    const bf16* __restrict__ A, const float* __restrict__ B,
    const float* __restrict__ bias, float* __restrict__ C, int K, int N) {
  __shared__ __align__(16) bf16 As[2][128][40];
  __shared__ __align__(16) bf16 Bt[2][128][40];
  const int tid  = threadIdx.x;
  const int wave = tid >> 5;
  const int m0 = blockIdx.y * 128;
  const int n0 = blockIdx.x * 128;
  const int wm = (wave >> 1) * 32;
  const int wn = (wave & 1) * 64;

  v8f acc[2][4];
#pragma unroll
  for (int i = 0; i < 2; ++i)
#pragma unroll
    for (int j = 0; j < 4; ++j)
#pragma unroll
      for (int e = 0; e < 8; ++e) acc[i][j][e] = 0.0f;

  // tile loaders -------------------------------------------------------
  auto issueA = [&](int buf, int k0) {               // async, no VGPR data
#pragma unroll
    for (int it = 0; it < 4; ++it) {
      int g = tid + it * 256;                        // 1024 groups of 4 bf16
      int r = g >> 3, c4 = (g & 7) * 4;
      async_b64(&As[buf][r][c4], A + (size_t)(m0 + r) * K + k0 + c4);
    }
  };
  float4 wreg[4];
  auto issueB = [&](int k0) {                        // global f32 loads only
#pragma unroll
    for (int it = 0; it < 4; ++it) {
      int g = tid + it * 256;
      int kr = g >> 5, c4 = (g & 31) * 4;
      wreg[it] = *(const float4*)(B + (size_t)(k0 + kr) * N + n0 + c4);
    }
    if (k0 + 64 < K) {                               // warm L2 two steps ahead
      int kr = tid >> 5, c4 = (tid & 31) * 4;
      __builtin_prefetch(B + (size_t)(k0 + 64 + kr) * N + n0 + c4, 0, 1);
    }
  };
  auto commitB = [&](int buf) {                      // cvt + LDS store (K-transposed)
#pragma unroll
    for (int it = 0; it < 4; ++it) {
      int g = tid + it * 256;
      int kr = g >> 5, c4 = (g & 31) * 4;
      Bt[buf][c4 + 0][kr] = f2bf(wreg[it].x);
      Bt[buf][c4 + 1][kr] = f2bf(wreg[it].y);
      Bt[buf][c4 + 2][kr] = f2bf(wreg[it].z);
      Bt[buf][c4 + 3][kr] = f2bf(wreg[it].w);
    }
  };

  // prologue: stage tile 0
  issueA(0, 0);
  issueB(0);
  commitB(0);
  async_wait0();
  __syncthreads();

  for (int k0 = 0; k0 < K; k0 += 32) {
    const int cur = (k0 >> 5) & 1;
    const int nxt = cur ^ 1;
    const bool more = (k0 + 32) < K;
    if (more) { issueA(nxt, k0 + 32); issueB(k0 + 32); }   // in flight during WMMA

    v16bf a0 = load_frag(&As[cur][wm][0], 40);
    v16bf a1 = load_frag(&As[cur][wm + 16][0], 40);
#pragma unroll
    for (int j = 0; j < 4; ++j) {
      v16bf b = load_frag(&Bt[cur][wn + j * 16][0], 40);
      acc[0][j] = __builtin_amdgcn_wmma_f32_16x16x32_bf16(
          false, a0, false, b, (short)0, acc[0][j], false, false);
      acc[1][j] = __builtin_amdgcn_wmma_f32_16x16x32_bf16(
          false, a1, false, b, (short)0, acc[1][j], false, false);
    }

    if (more) commitB(nxt);
    async_wait0();
    __syncthreads();
  }

  const int lane = tid & 31;
  const int cN = lane & 15, rH = (lane >> 4) * 8;
#pragma unroll
  for (int i = 0; i < 2; ++i)
#pragma unroll
    for (int j = 0; j < 4; ++j) {
      int col = n0 + wn + j * 16 + cN;
      float bv = bias ? bias[col] : 0.0f;
      int rb = m0 + wm + i * 16 + rH;
#pragma unroll
      for (int v = 0; v < 8; ++v)
        C[(size_t)(rb + v) * N + col] = acc[i][j][v] + bv;
    }
}

// ---------- Fused gated FFN GEMM: act = gelu(h@Wg) * (h@W1), bf16 out ----------
// 128x64 block tile, dual accumulators, 2x2 tiles per wave, same pipeline.

__global__ __launch_bounds__(256) void gemm_gated_kernel(
    const bf16* __restrict__ A, const float* __restrict__ Bg,
    const float* __restrict__ Bu, bf16* __restrict__ Act, int K, int N) {
  __shared__ __align__(16) bf16 As[2][128][40];
  __shared__ __align__(16) bf16 Btg[2][64][40];
  __shared__ __align__(16) bf16 Btu[2][64][40];
  const int tid  = threadIdx.x;
  const int wave = tid >> 5;
  const int m0 = blockIdx.y * 128;
  const int n0 = blockIdx.x * 64;
  const int wm = (wave >> 1) * 32;
  const int wn = (wave & 1) * 32;

  v8f ag[2][2], au[2][2];
#pragma unroll
  for (int i = 0; i < 2; ++i)
#pragma unroll
    for (int j = 0; j < 2; ++j)
#pragma unroll
      for (int e = 0; e < 8; ++e) { ag[i][j][e] = 0.0f; au[i][j][e] = 0.0f; }

  auto issueA = [&](int buf, int k0) {
#pragma unroll
    for (int it = 0; it < 4; ++it) {
      int g = tid + it * 256;
      int r = g >> 3, c4 = (g & 7) * 4;
      async_b64(&As[buf][r][c4], A + (size_t)(m0 + r) * K + k0 + c4);
    }
  };
  float4 wg[2], wu[2];
  auto issueB = [&](int k0) {
#pragma unroll
    for (int it = 0; it < 2; ++it) {
      int g = tid + it * 256;                        // 512 groups of 4 f32
      int kr = g >> 4, c4 = (g & 15) * 4;
      wg[it] = *(const float4*)(Bg + (size_t)(k0 + kr) * N + n0 + c4);
      wu[it] = *(const float4*)(Bu + (size_t)(k0 + kr) * N + n0 + c4);
    }
    if (k0 + 64 < K) {
      int kr = tid >> 3, c4 = (tid & 7) * 4;         // 256 thr cover 32x32 region
      __builtin_prefetch(Bg + (size_t)(k0 + 64 + kr) * N + n0 + c4, 0, 1);
      __builtin_prefetch(Bu + (size_t)(k0 + 64 + kr) * N + n0 + c4, 0, 1);
    }
  };
  auto commitB = [&](int buf) {
#pragma unroll
    for (int it = 0; it < 2; ++it) {
      int g = tid + it * 256;
      int kr = g >> 4, c4 = (g & 15) * 4;
      Btg[buf][c4 + 0][kr] = f2bf(wg[it].x); Btg[buf][c4 + 1][kr] = f2bf(wg[it].y);
      Btg[buf][c4 + 2][kr] = f2bf(wg[it].z); Btg[buf][c4 + 3][kr] = f2bf(wg[it].w);
      Btu[buf][c4 + 0][kr] = f2bf(wu[it].x); Btu[buf][c4 + 1][kr] = f2bf(wu[it].y);
      Btu[buf][c4 + 2][kr] = f2bf(wu[it].z); Btu[buf][c4 + 3][kr] = f2bf(wu[it].w);
    }
  };

  issueA(0, 0);
  issueB(0);
  commitB(0);
  async_wait0();
  __syncthreads();

  for (int k0 = 0; k0 < K; k0 += 32) {
    const int cur = (k0 >> 5) & 1;
    const int nxt = cur ^ 1;
    const bool more = (k0 + 32) < K;
    if (more) { issueA(nxt, k0 + 32); issueB(k0 + 32); }

    v16bf a0 = load_frag(&As[cur][wm][0], 40);
    v16bf a1 = load_frag(&As[cur][wm + 16][0], 40);
#pragma unroll
    for (int j = 0; j < 2; ++j) {
      v16bf bg = load_frag(&Btg[cur][wn + j * 16][0], 40);
      v16bf bu = load_frag(&Btu[cur][wn + j * 16][0], 40);
      ag[0][j] = __builtin_amdgcn_wmma_f32_16x16x32_bf16(
          false, a0, false, bg, (short)0, ag[0][j], false, false);
      ag[1][j] = __builtin_amdgcn_wmma_f32_16x16x32_bf16(
          false, a1, false, bg, (short)0, ag[1][j], false, false);
      au[0][j] = __builtin_amdgcn_wmma_f32_16x16x32_bf16(
          false, a0, false, bu, (short)0, au[0][j], false, false);
      au[1][j] = __builtin_amdgcn_wmma_f32_16x16x32_bf16(
          false, a1, false, bu, (short)0, au[1][j], false, false);
    }

    if (more) commitB(nxt);
    async_wait0();
    __syncthreads();
  }

  const int lane = tid & 31;
  const int cN = lane & 15, rH = (lane >> 4) * 8;
#pragma unroll
  for (int i = 0; i < 2; ++i)
#pragma unroll
    for (int j = 0; j < 2; ++j) {
      int col = n0 + wn + j * 16 + cN;
      int rb = m0 + wm + i * 16 + rH;
#pragma unroll
      for (int v = 0; v < 8; ++v)
        Act[(size_t)(rb + v) * N + col] =
            f2bf(gelu_tanh(ag[i][j][v]) * au[i][j][v]);
    }
}

// ---------- RMSNorm (row) -> bf16 activations ----------

__global__ __launch_bounds__(256) void rmsnorm_bf16_kernel(
    const float* __restrict__ X, const float* __restrict__ sc,
    bf16* __restrict__ O) {
  __shared__ float red[256];
  const int row = blockIdx.x, t = threadIdx.x;
  const float* x = X + (size_t)row * D_;
  float s = 0.0f;
  for (int i = t; i < D_; i += 256) { float v = x[i]; s += v * v; }
  red[t] = s; __syncthreads();
  for (int o = 128; o > 0; o >>= 1) { if (t < o) red[t] += red[t + o]; __syncthreads(); }
  float r = rsqrtf(red[0] / (float)D_ + 1e-6f);
  for (int i = t; i < D_; i += 256)
    O[(size_t)row * D_ + i] = f2bf(x[i] * r * (1.0f + sc[i]));
}

// ---------- x += rmsnorm(t, sc); optional row zeroing by pos_mask ----------

__global__ __launch_bounds__(256) void addnorm_kernel(
    float* __restrict__ X, const float* __restrict__ T,
    const float* __restrict__ sc, const unsigned char* __restrict__ pm) {
  __shared__ float red[256];
  const int row = blockIdx.x, t = threadIdx.x;
  const float* tr = T + (size_t)row * D_;
  float s = 0.0f;
  for (int i = t; i < D_; i += 256) { float v = tr[i]; s += v * v; }
  red[t] = s; __syncthreads();
  for (int o = 128; o > 0; o >>= 1) { if (t < o) red[t] += red[t + o]; __syncthreads(); }
  float r = rsqrtf(red[0] / (float)D_ + 1e-6f);
  const bool zero = (pm != nullptr) && (pm[row] == 0);
  for (int i = t; i < D_; i += 256) {
    size_t idx = (size_t)row * D_ + i;
    float v = X[idx] + tr[i] * r * (1.0f + sc[i]);
    X[idx] = zero ? 0.0f : v;
  }
}

// ---------- pos_mask[row] = any(attn_mask[row, :]) ----------

__global__ __launch_bounds__(256) void posmask_kernel(
    const unsigned char* __restrict__ m, unsigned char* __restrict__ pm) {
  __shared__ int red[256];
  const int row = blockIdx.x, t = threadIdx.x;
  const unsigned int* mr = (const unsigned int*)(m + (size_t)row * S_);
  int any = 0;
  for (int i = t; i < S_ / 4; i += 256) any |= (mr[i] != 0u);
  red[t] = any; __syncthreads();
  for (int o = 128; o > 0; o >>= 1) { if (t < o) red[t] |= red[t + o]; __syncthreads(); }
  if (t == 0) pm[row] = (unsigned char)(red[0] ? 1 : 0);
}

// ---------- per-(s,h): q/k head RMSNorm + RoPE, v passthrough, -> bf16 [H][S][64] ----------

__global__ __launch_bounds__(64) void qkrope_kernel(
    const float* __restrict__ qf, const float* __restrict__ kf,
    const float* __restrict__ vf, const float* __restrict__ qn,
    const float* __restrict__ kn, const int* __restrict__ pos,
    bf16* __restrict__ qb, bf16* __restrict__ kb, bf16* __restrict__ vb) {
  const int h = blockIdx.x, s = blockIdx.y, d = threadIdx.x;
  __shared__ float red[64];
  __shared__ float tq[64], tk[64];
  size_t ii = (size_t)s * (H_ * HD_) + h * HD_ + d;
  float qv = qf[ii], kv = kf[ii], vv = vf[ii];

  red[d] = qv * qv; __syncthreads();
  for (int o = 32; o > 0; o >>= 1) { if (d < o) red[d] += red[d + o]; __syncthreads(); }
  float qr = rsqrtf(red[0] / (float)HD_ + 1e-6f);
  __syncthreads();
  red[d] = kv * kv; __syncthreads();
  for (int o = 32; o > 0; o >>= 1) { if (d < o) red[d] += red[d + o]; __syncthreads(); }
  float kr = rsqrtf(red[0] / (float)HD_ + 1e-6f);
  __syncthreads();

  tq[d] = qv * qr * (1.0f + qn[d]);
  tk[d] = kv * kr * (1.0f + kn[d]);
  __syncthreads();

  float p = (float)pos[s];
  int i = (d < 32) ? d : d - 32;
  float ts = powf(10000.0f, (2.0f * (float)i) / (float)HD_);
  float ang = p / ts;
  float sn = sinf(ang), cs = cosf(ang);
  float oq, ok;
  if (d < 32) { oq = tq[d] * cs - tq[d + 32] * sn; ok = tk[d] * cs - tk[d + 32] * sn; }
  else        { oq = tq[d] * cs + tq[d - 32] * sn; ok = tk[d] * cs + tk[d - 32] * sn; }

  size_t ob = ((size_t)h * S_ + s) * HD_ + d;
  qb[ob] = f2bf(oq); kb[ob] = f2bf(ok); vb[ob] = f2bf(vv);
}

// ---------- flash attention: 1 WG = (head, 64 Q rows), 4 waves, each wave = 16 rows ----------

__global__ __launch_bounds__(128) void attn_kernel(
    const bf16* __restrict__ qb, const bf16* __restrict__ kb,
    const bf16* __restrict__ vb, const unsigned char* __restrict__ mask,
    bf16* __restrict__ out) {
  __shared__ __align__(16) bf16 Kt[64][72];   // [seq n][hd k]
  __shared__ __align__(16) bf16 Vt[64][72];   // [hd n][seq k]
  __shared__ __align__(16) bf16 Ps[64][72];   // [q row][seq col]
  const int tid = threadIdx.x;
  const int wave = tid >> 5, lane = tid & 31;
  const int head = blockIdx.y;
  const int q0 = blockIdx.x * 64;
  const int cN = lane & 15, rH = (lane >> 4) * 8;

  const bf16* qbase = qb + ((size_t)head * S_ + q0 + wave * 16) * HD_;
  v16bf aQ0 = load_frag(qbase, HD_);
  v16bf aQ1 = load_frag(qbase + 32, HD_);

  v8f o[4];
  float mr[8], lr[8];
#pragma unroll
  for (int j = 0; j < 4; ++j)
#pragma unroll
    for (int e = 0; e < 8; ++e) o[j][e] = 0.0f;
#pragma unroll
  for (int v = 0; v < 8; ++v) { mr[v] = -1e30f; lr[v] = 0.0f; }

  for (int kt = 0; kt < S_ / 64; ++kt) {
    // K tile: async copy, already [seq][hd] == Bt[n][k] for Q@K^T
#pragma unroll
    for (int it = 0; it < 8; ++it) {
      int g = tid + it * 128;
      int r = g >> 4, c4 = (g & 15) * 4;
      async_b64(&Kt[r][c4],
                kb + ((size_t)head * S_ + kt * 64 + r) * HD_ + c4);
    }
    // V tile transposed through VGPRs: Vt[hd][seq]
#pragma unroll
    for (int it = 0; it < 8; ++it) {
      int g = tid + it * 128;
      int r = g >> 4, c4 = (g & 15) * 4;
      const bf16* vp = vb + ((size_t)head * S_ + kt * 64 + r) * HD_ + c4;
      Vt[c4 + 0][r] = vp[0]; Vt[c4 + 1][r] = vp[1];
      Vt[c4 + 2][r] = vp[2]; Vt[c4 + 3][r] = vp[3];
    }
    async_wait0();
    __syncthreads();

    // scores: 16 x 64 per wave
    v8f sc[4];
#pragma unroll
    for (int nt = 0; nt < 4; ++nt) {
      v8f s;
#pragma unroll
      for (int e = 0; e < 8; ++e) s[e] = 0.0f;
      v16bf b0 = load_frag(&Kt[nt * 16][0], 72);
      v16bf b1 = load_frag(&Kt[nt * 16][32], 72);
      s = __builtin_amdgcn_wmma_f32_16x16x32_bf16(false, aQ0, false, b0, (short)0, s, false, false);
      s = __builtin_amdgcn_wmma_f32_16x16x32_bf16(false, aQ1, false, b1, (short)0, s, false, false);
      sc[nt] = s;
    }

    // scale + mask
#pragma unroll
    for (int nt = 0; nt < 4; ++nt)
#pragma unroll
      for (int v = 0; v < 8; ++v) {
        int rq = q0 + wave * 16 + rH + v;
        int ck = kt * 64 + nt * 16 + cN;
        float val = sc[nt][v] * 0.125f;     // 1/sqrt(64)
        if (!mask[(size_t)rq * S_ + ck]) val = -1e30f;
        sc[nt][v] = val;
      }

    // online softmax (rows live across 16-lane halves; vgpr v = row)
    float alpha[8];
#pragma unroll
    for (int v = 0; v < 8; ++v) {
      float mx = fmaxf(fmaxf(sc[0][v], sc[1][v]), fmaxf(sc[2][v], sc[3][v]));
#pragma unroll
      for (int off = 1; off < 16; off <<= 1) mx = fmaxf(mx, __shfl_xor(mx, off, 32));
      float mnew = fmaxf(mr[v], mx);
      alpha[v] = __expf(mr[v] - mnew);
      float rs = 0.0f;
#pragma unroll
      for (int nt = 0; nt < 4; ++nt) {
        float p = __expf(sc[nt][v] - mnew);
        sc[nt][v] = p; rs += p;
      }
#pragma unroll
      for (int off = 1; off < 16; off <<= 1) rs += __shfl_xor(rs, off, 32);
      lr[v] = lr[v] * alpha[v] + rs;
      mr[v] = mnew;
    }

    // P -> LDS (bf16), rescale O accumulators
#pragma unroll
    for (int nt = 0; nt < 4; ++nt)
#pragma unroll
      for (int v = 0; v < 8; ++v)
        Ps[wave * 16 + rH + v][nt * 16 + cN] = f2bf(sc[nt][v]);
#pragma unroll
    for (int j = 0; j < 4; ++j)
#pragma unroll
      for (int v = 0; v < 8; ++v) o[j][v] *= alpha[v];
    __syncthreads();

    // O += P @ V
    v16bf aP0 = load_frag(&Ps[wave * 16][0], 72);
    v16bf aP1 = load_frag(&Ps[wave * 16][32], 72);
#pragma unroll
    for (int j = 0; j < 4; ++j) {
      v16bf b0 = load_frag(&Vt[j * 16][0], 72);
      v16bf b1 = load_frag(&Vt[j * 16][32], 72);
      o[j] = __builtin_amdgcn_wmma_f32_16x16x32_bf16(false, aP0, false, b0, (short)0, o[j], false, false);
      o[j] = __builtin_amdgcn_wmma_f32_16x16x32_bf16(false, aP1, false, b1, (short)0, o[j], false, false);
    }
    __syncthreads();
  }

  // normalize + write bf16 [S][H*HD]
#pragma unroll
  for (int j = 0; j < 4; ++j)
#pragma unroll
    for (int v = 0; v < 8; ++v) {
      int rq = q0 + wave * 16 + rH + v;
      int cd = j * 16 + cN;
      out[(size_t)rq * (H_ * HD_) + head * HD_ + cd] = f2bf(o[j][v] / lr[v]);
    }
}

// ---------- orchestration ----------

extern "C" void kernel_launch(void* const* d_in, const int* in_sizes, int n_in,
                              void* d_out, int out_size, void* d_ws, size_t ws_size,
                              hipStream_t stream) {
  const float* qkv  = (const float*)d_in[0];
  const unsigned char* amask = (const unsigned char*)d_in[1];
  const int*   positions = (const int*)d_in[2];
  const float* ln1 = (const float*)d_in[3];
  const float* Wq = (const float*)d_in[4];  const float* bq = (const float*)d_in[5];
  const float* Wk = (const float*)d_in[6];  const float* bk = (const float*)d_in[7];
  const float* Wv = (const float*)d_in[8];  const float* bv = (const float*)d_in[9];
  const float* qn = (const float*)d_in[10]; const float* kn = (const float*)d_in[11];
  const float* Wo = (const float*)d_in[12]; const float* bo = (const float*)d_in[13];
  const float* pa = (const float*)d_in[14];
  const float* ln2 = (const float*)d_in[15];
  const float* Wg = (const float*)d_in[16];
  const float* W1 = (const float*)d_in[17];
  const float* W2 = (const float*)d_in[18];
  const float* pf = (const float*)d_in[19];

  char* ws = (char*)d_ws;
  size_t off = 0;
  auto alloc = [&](size_t bytes) -> void* {
    void* p = ws + off;
    off += (bytes + 255) & ~(size_t)255;
    return p;
  };
  float* x   = (float*)alloc((size_t)S_ * D_ * 4);
  bf16*  hb  = (bf16*)alloc((size_t)S_ * D_ * 2);
  float* qf  = (float*)alloc((size_t)S_ * D_ * 4);
  float* kf  = (float*)alloc((size_t)S_ * D_ * 4);
  float* vf  = (float*)alloc((size_t)S_ * D_ * 4);
  bf16*  qbb = (bf16*)alloc((size_t)S_ * D_ * 2);
  bf16*  kbb = (bf16*)alloc((size_t)S_ * D_ * 2);
  bf16*  vbb = (bf16*)alloc((size_t)S_ * D_ * 2);
  bf16*  ao  = (bf16*)alloc((size_t)S_ * D_ * 2);
  float* t1  = (float*)alloc((size_t)S_ * D_ * 4);
  bf16*  act = (bf16*)alloc((size_t)S_ * F_ * 2);
  unsigned char* pm = (unsigned char*)alloc(S_);

  hipMemcpyAsync(x, qkv, (size_t)S_ * D_ * 4, hipMemcpyDeviceToDevice, stream);
  posmask_kernel<<<S_, 256, 0, stream>>>(amask, pm);

  for (int l = 0; l < L_; ++l) {
    rmsnorm_bf16_kernel<<<S_, 256, 0, stream>>>(x, ln1 + (size_t)l * D_, hb);
    gemm_bf16_kernel<<<dim3(D_ / 128, S_ / 128), 256, 0, stream>>>(
        hb, Wq + (size_t)l * D_ * D_, bq + (size_t)l * D_, qf, D_, D_);
    gemm_bf16_kernel<<<dim3(D_ / 128, S_ / 128), 256, 0, stream>>>(
        hb, Wk + (size_t)l * D_ * D_, bk + (size_t)l * D_, kf, D_, D_);
    gemm_bf16_kernel<<<dim3(D_ / 128, S_ / 128), 256, 0, stream>>>(
        hb, Wv + (size_t)l * D_ * D_, bv + (size_t)l * D_, vf, D_, D_);
    qkrope_kernel<<<dim3(H_, S_), 64, 0, stream>>>(
        qf, kf, vf, qn + (size_t)l * HD_, kn + (size_t)l * HD_, positions,
        qbb, kbb, vbb);
    attn_kernel<<<dim3(S_ / 64, H_), 128, 0, stream>>>(qbb, kbb, vbb, amask, ao);
    gemm_bf16_kernel<<<dim3(D_ / 128, S_ / 128), 256, 0, stream>>>(
        ao, Wo + (size_t)l * D_ * D_, bo + (size_t)l * D_, t1, D_, D_);
    addnorm_kernel<<<S_, 256, 0, stream>>>(x, t1, pa + (size_t)l * D_, nullptr);
    rmsnorm_bf16_kernel<<<S_, 256, 0, stream>>>(x, ln2 + (size_t)l * D_, hb);
    gemm_gated_kernel<<<dim3(F_ / 64, S_ / 128), 256, 0, stream>>>(
        hb, Wg + (size_t)l * D_ * F_, W1 + (size_t)l * D_ * F_, act, D_, F_);
    gemm_bf16_kernel<<<dim3(D_ / 128, S_ / 128), 256, 0, stream>>>(
        act, W2 + (size_t)l * F_ * D_, nullptr, t1, F_, D_);
    addnorm_kernel<<<S_, 256, 0, stream>>>(x, t1, pf + (size_t)l * D_, pm);
  }

  hipMemcpyAsync(d_out, x, (size_t)S_ * D_ * 4, hipMemcpyDeviceToDevice, stream);
}
